// TrajectoryPredictor_78288663871798
// MI455X (gfx1250) — matrix operand order repaired and checked by reading, not compiled, not run on hardware
//
#include <hip/hip_runtime.h>
#include <hip/hip_bf16.h>

typedef __bf16 bf16_t;
typedef __attribute__((ext_vector_type(8)))  bf16_t v8bf;
typedef __attribute__((ext_vector_type(16))) bf16_t v16bf;
typedef __attribute__((ext_vector_type(8)))  float  v8f;

#define B_   128
#define T_   256
#define I_   4
#define H_   64
#define G_   256   // 4*H, gate dim (i,f,g,o)
#define L_   32
#define WS_  136   // padded row stride (bf16) for weights & h-concat (bank-shift 4/row)
#define CS_  68    // padded row stride (f32) for cell state
#define RING 4

__device__ __forceinline__ float fast_sig(float x) {
  float e = __builtin_amdgcn_exp2f(-1.442695040f * x);
  return __builtin_amdgcn_rcpf(1.0f + e);
}
__device__ __forceinline__ float fast_tanh(float x) {
  float e = __builtin_amdgcn_exp2f(2.885390082f * x);   // 2^(2x/ln2) = e^(2x)
  return 1.0f - 2.0f * __builtin_amdgcn_rcpf(e + 1.0f);
}

__global__ void lstm_init_flags(int* prog) {
  if (threadIdx.x < L_) prog[threadIdx.x] = 0;
}

// One workgroup per layer; wavefront pipeline over (layer, t) via global ring buffers.
__global__ __launch_bounds__(256, 1)
void lstm_pipeline(const float* __restrict__ x,
                   const int*   __restrict__ lengths,
                   const float* __restrict__ w_ih0,
                   const float* __restrict__ w_ih,
                   const float* __restrict__ w_hh,
                   const float* __restrict__ b_ih,
                   const float* __restrict__ b_hh,
                   const float* __restrict__ fc_w,
                   const float* __restrict__ fc_b,
                   float*       __restrict__ out,
                   int*         __restrict__ prog,
                   bf16_t*      __restrict__ hring) {
  const int l      = blockIdx.x;        // layer id, 0..31
  const int tid    = threadIdx.x;       // 256 threads = 8 waves
  const int lane   = tid & 31;
  const int mt     = tid >> 5;          // wave id -> batch M-tile (16 rows)
  const int laneN  = lane & 15;
  const int hiHalf = lane >> 4;         // 0: lanes 0-15, 1: lanes 16-31

  extern __shared__ char smem[];
  bf16_t* sW    = (bf16_t*)smem;                  // [G_][WS_]  W_cat = [w_in | w_hh], bf16
  bf16_t* sH    = sW + G_ * WS_;                  // [B_][WS_]  cols 0..63 = input, 64..127 = own h
  float*  sC    = (float*)(sH + B_ * WS_);        // [B_][CS_]  cell state
  float*  sBias = sC + B_ * CS_;                  // [G_]
  int*    sLen  = (int*)(sBias + G_);             // [B_]

  // ---- one-time LDS setup: weights (f32 -> bf16), bias sums, state = 0 ----
  for (int i = tid; i < G_ * WS_; i += 256) {
    int g = i / WS_, k = i - g * WS_;
    float w = 0.f;
    if (k < 64) {
      if (l == 0) { if (k < I_) w = w_ih0[g * I_ + k]; }
      else        { w = w_ih[((size_t)(l - 1) * G_ + g) * H_ + k]; }
    } else if (k < 128) {
      w = w_hh[((size_t)l * G_ + g) * H_ + (k - 64)];
    }
    sW[i] = (bf16_t)w;
  }
  for (int i = tid; i < B_ * WS_; i += 256) sH[i] = (bf16_t)0.f;
  for (int i = tid; i < B_ * CS_; i += 256) sC[i] = 0.f;
  for (int i = tid; i < G_; i += 256) sBias[i] = b_ih[(size_t)l * G_ + i] + b_hh[(size_t)l * G_ + i];
  for (int i = tid; i < B_; i += 256) sLen[i] = lengths[i];
  __syncthreads();

  const bf16_t* ringIn  = hring + (size_t)(l - 1) * RING * B_ * H_;
  bf16_t*       ringOut = hring + (size_t)l       * RING * B_ * H_;

  // A-fragment base: 16-bit A 16x32 layout -> lane holds row M = mt*16 + lane%16;
  // K runs: [kt*32 + (hi?8:0) .. +7] and [kt*32 + (hi?24:16) .. +7]
  const bf16_t* aRow = sH + (mt * 16 + laneN) * WS_;
  const int aO = hiHalf * 8;

  for (int t = 0; t < T_; ++t) {
    // ---- acquire input tile and fill sH cols 0..63 ----
    if (l > 0) {
      while (__hip_atomic_load(&prog[l - 1], __ATOMIC_ACQUIRE, __HIP_MEMORY_SCOPE_AGENT) <= t)
        __builtin_amdgcn_s_sleep(1);
      const bf16_t* src = ringIn + (size_t)(t & (RING - 1)) * B_ * H_;
      int row = tid >> 1, part = tid & 1;
      const uint4* s4 = (const uint4*)(src + row * H_ + part * 32);
      uint4* d4 = (uint4*)(sH + row * WS_ + part * 32);
      d4[0] = s4[0]; d4[1] = s4[1]; d4[2] = s4[2]; d4[3] = s4[3];
    } else {
      if (tid < B_) {
        const float* xp = x + ((size_t)tid * T_ + t) * I_;
        bf16_t* dp = sH + tid * WS_;
        dp[0] = (bf16_t)xp[0]; dp[1] = (bf16_t)xp[1];
        dp[2] = (bf16_t)xp[2]; dp[3] = (bf16_t)xp[3];
      }
    }
    // back-pressure: don't overwrite a ring slot the consumer hasn't read
    if (l < L_ - 1 && t >= RING) {
      while (__hip_atomic_load(&prog[l + 1], __ATOMIC_ACQUIRE, __HIP_MEMORY_SCOPE_AGENT) < t - RING + 1)
        __builtin_amdgcn_s_sleep(1);
    }
    __syncthreads();

    // ---- preload the 4 A-fragments for this step (wave-private rows of sH) ----
    v16bf Af[4];
    #pragma unroll
    for (int kt = 0; kt < 4; ++kt) {
      const bf16_t* ap = aRow + kt * 32 + aO;
      v8bf a0 = *(const v8bf*)(ap);
      v8bf a1 = *(const v8bf*)(ap + 16);
      Af[kt] = __builtin_shufflevector(a0, a1, 0,1,2,3,4,5,6,7,8,9,10,11,12,13,14,15);
    }

    // ---- 4 gate-groups: tiles {g4, g4+4, g4+8, g4+12} = i/f/g/o for hidden
    //      units hh in [g4*16, g4*16+16). Only 4 accumulators live at a time. ----
    bf16_t* gout = ringOut + (size_t)(t & (RING - 1)) * B_ * H_;
    #pragma unroll
    for (int g4 = 0; g4 < 4; ++g4) {
      v8f acc[4];
      #pragma unroll
      for (int j = 0; j < 4; ++j) {
        v8f z;
        #pragma unroll
        for (int e = 0; e < 8; ++e) z[e] = 0.f;
        acc[j] = z;
      }
      #pragma unroll
      for (int kt = 0; kt < 4; ++kt) {
        #pragma unroll
        for (int j = 0; j < 4; ++j) {
          int n = g4 + 4 * j;
          // B 32x16 layout: lane holds col N = n*16 + lane%16; K runs: [kt*32 + (hi?8:0) .. +7] and [kt*32 + (hi?24:16) .. +7]
          const bf16_t* bp = sW + (n * 16 + laneN) * WS_ + kt * 32 + hiHalf * 8;
          v8bf b0 = *(const v8bf*)(bp);
          v8bf b1 = *(const v8bf*)(bp + 16);
          v16bf Bf = __builtin_shufflevector(b0, b1, 0,1,2,3,4,5,6,7,8,9,10,11,12,13,14,15);
          acc[j] = __builtin_amdgcn_wmma_f32_16x16x32_bf16(
              false, Af[kt], false, Bf, (short)0, acc[j], false, false);
        }
      }

      // ---- elementwise LSTM cell update for this hidden group ----
      const int hh = g4 * 16 + laneN;
      const float bi = sBias[hh];
      const float bff = sBias[64 + hh];
      const float bg = sBias[128 + hh];
      const float bo = sBias[192 + hh];
      #pragma unroll
      for (int r = 0; r < 8; ++r) {
        int row = mt * 16 + r + hiHalf * 8;    // C/D layout: VGPR r -> M = r (+8 hi half)
        bool valid = t < sLen[row];
        float iv = acc[0][r] + bi;
        float fv = acc[1][r] + bff;
        float gv = acc[2][r] + bg;
        float ov = acc[3][r] + bo;
        float cOld = sC[row * CS_ + hh];
        float cNew = fast_sig(fv) * cOld + fast_sig(iv) * fast_tanh(gv);
        float hNew = fast_sig(ov) * fast_tanh(cNew);
        bf16_t hOldb = sH[row * WS_ + 64 + hh];
        float  cOut = valid ? cNew : cOld;           // freeze past sequence end
        bf16_t hOut = valid ? (bf16_t)hNew : hOldb;
        sC[row * CS_ + hh] = cOut;
        sH[row * WS_ + 64 + hh] = hOut;
        if (l < L_ - 1) gout[row * H_ + hh] = hOut;  // publish to next layer
      }
    }
    __threadfence();       // release our global h stores
    __syncthreads();       // all waves done; safe to refill input cols next step
    if (tid == 0)
      __hip_atomic_store(&prog[l], t + 1, __ATOMIC_RELEASE, __HIP_MEMORY_SCOPE_AGENT);
  }

  // ---- final FC head (layer 31 block only): out[128,4] = hT @ fc_w^T + fc_b ----
  if (l == L_ - 1) {
    __syncthreads();
    for (int o = tid; o < B_ * 4; o += 256) {
      int row = o >> 2, j = o & 3;
      float s = fc_b[j];
      const bf16_t* hp = sH + row * WS_ + 64;
      #pragma unroll
      for (int k = 0; k < H_; ++k) s += (float)hp[k] * fc_w[j * H_ + k];
      out[o] = s;
    }
  }
}

extern "C" void kernel_launch(void* const* d_in, const int* in_sizes, int n_in,
                              void* d_out, int out_size, void* d_ws, size_t ws_size,
                              hipStream_t stream) {
  const float* x       = (const float*)d_in[0];
  const int*   lengths = (const int*)  d_in[1];
  const float* w_ih0   = (const float*)d_in[2];
  const float* w_ih    = (const float*)d_in[3];
  const float* w_hh    = (const float*)d_in[4];
  const float* b_ih    = (const float*)d_in[5];
  const float* b_hh    = (const float*)d_in[6];
  const float* fc_w    = (const float*)d_in[7];
  const float* fc_b    = (const float*)d_in[8];
  float* out = (float*)d_out;

  int*    prog  = (int*)d_ws;                          // 32 progress counters
  bf16_t* hring = (bf16_t*)((char*)d_ws + 256);        // [L_][RING][B_][H_] bf16 (~2 MB)

  lstm_init_flags<<<dim3(1), dim3(32), 0, stream>>>(prog);

  size_t smem = (size_t)(G_ * WS_ + B_ * WS_) * sizeof(bf16_t)
              + (size_t)(B_ * CS_ + G_) * sizeof(float)
              + (size_t)B_ * sizeof(int);              // ~140.8 KB < 320 KB/WGP
  lstm_pipeline<<<dim3(L_), dim3(256), smem, stream>>>(
      x, lengths, w_ih0, w_ih, w_hh, b_ih, b_hh, fc_w, fc_b, out, prog, hring);
}
